// GCN_Model_sps_88759794139180
// MI455X (gfx1250) — matrix operand order repaired and checked by reading, not compiled, not run on hardware
//
#include <hip/hip_runtime.h>
#include <hip/hip_bf16.h>
#include <math.h>

// ---------------------------------------------------------------------------
// CDNA5 (gfx1250) GCN forward: wave32, WMMA bf16 16x16x32, f32 accumulate.
// ---------------------------------------------------------------------------

typedef __attribute__((ext_vector_type(16))) __bf16 v16bf;
typedef __attribute__((ext_vector_type(8)))  float  v8f;
typedef __attribute__((ext_vector_type(4)))  float  fvec4;   // native vector (nontemporal-ok)

union Frag {
    v16bf v;
    uint4 q[2];
};

__device__ __forceinline__ unsigned int f32_to_bf16_bits(float f) {
    unsigned int u = __float_as_uint(f);
    u += 0x7FFFu + ((u >> 16) & 1u);  // round-to-nearest-even
    return u >> 16;
}
__device__ __forceinline__ unsigned int pack_bf16x2(float lo, float hi) {
    return f32_to_bf16_bits(lo) | (f32_to_bf16_bits(hi) << 16);
}

// ---------------------------------------------------------------------------
// Row sums: D2[row] = sum_j tilde[row][j].  One block per row, float4 loads.
// ---------------------------------------------------------------------------
__global__ __launch_bounds__(256)
void rowsum_kernel(const float* __restrict__ tilde, float* __restrict__ D2, int n) {
    __shared__ float red[256];
    const fvec4* p = (const fvec4*)(tilde + (size_t)blockIdx.x * n);
    const int n4 = n >> 2;
    float acc = 0.0f;
    for (int j = threadIdx.x; j < n4; j += 256) {
        fvec4 v = p[j];
        acc += (v[0] + v[1]) + (v[2] + v[3]);
    }
    red[threadIdx.x] = acc;
    __syncthreads();
    for (int s = 128; s > 0; s >>= 1) {
        if (threadIdx.x < s) red[threadIdx.x] += red[threadIdx.x + s];
        __syncthreads();
    }
    if (threadIdx.x == 0) D2[blockIdx.x] = red[0];
}

// ---------------------------------------------------------------------------
// Column-sum partials over row chunks (deterministic, no atomics).
// ---------------------------------------------------------------------------
__global__ __launch_bounds__(256)
void colsum_part_kernel(const float* __restrict__ tilde, float* __restrict__ part,
                        int n, int rowsPer) {
    const int col = blockIdx.x * 256 + threadIdx.x;
    const int r0  = blockIdx.y * rowsPer;
    float acc = 0.0f;
    for (int r = 0; r < rowsPer; ++r) acc += tilde[(size_t)(r0 + r) * n + col];
    part[(size_t)blockIdx.y * n + col] = acc;
}

// Final col reduce (fixed order) fused with scale[j] = sqrt(D1[j])*sqrt(D2[j]).
__global__ __launch_bounds__(256)
void colsum_scale_kernel(const float* __restrict__ part, const float* __restrict__ D2,
                         float* __restrict__ scl, int n, int nparts) {
    const int col = blockIdx.x * 256 + threadIdx.x;
    float d1 = 0.0f;
    for (int p = 0; p < nparts; ++p) d1 += part[(size_t)p * n + col];
    scl[col] = sqrtf(d1) * sqrtf(D2[col]);
}

// Fixed-order sum of nz partial planes (split-K reduce).
__global__ __launch_bounds__(256)
void splitk_reduce_kernel(const float* __restrict__ part, float* __restrict__ out,
                          int count, int nz, size_t stride) {
    const int i = blockIdx.x * 256 + threadIdx.x;
    if (i >= count) return;
    float s = 0.0f;
    for (int z = 0; z < nz; ++z) s += part[(size_t)z * stride + i];
    out[i] = s;
}

// ---------------------------------------------------------------------------
// GEMM: C[M x Ncols] = op(A_scaled[M x K] @ B[K x Ncols] + bias)
// Block 256 thr (8 waves), tile 128x128, wave tile 32x64 (2x4 WMMA accums).
// Register double-buffered global->LDS staging; bf16 packed into u32 LDS words.
// gridDim.z > 1 => deterministic split-K: plane z covers K-chunk z, writes to
// Cmat + z*strideZ (bias/relu must be off in that case).
// ---------------------------------------------------------------------------
template<bool HAS_SCALE, bool HAS_BIAS, bool RELU, bool NGUARD>
__global__ __launch_bounds__(256)
void gemm_bf16_wmma(const float* __restrict__ A, int lda,
                    const float* __restrict__ B, int ldb,
                    float* __restrict__ Cmat, int ldc,
                    int Ncols, int K, int kChunk, long long strideZ,
                    const float* __restrict__ colScale,
                    const float* __restrict__ bias)
{
    __shared__ unsigned int Asm[128][16];   // [m][k/2]  packed bf16x2
    __shared__ unsigned int Bsm[128][16];   // [n][k/2]  packed bf16x2 (transposed)

    const int t    = threadIdx.x;
    const int m0   = blockIdx.y * 128;
    const int n0   = blockIdx.x * 128;
    const int kBeg = blockIdx.z * kChunk;
    const int kEnd = (kBeg + kChunk < K) ? (kBeg + kChunk) : K;

    const int lane = t & 31;
    const int w    = t >> 5;
    const int wm   = (w & 3) * 32;
    const int wn   = (w >> 2) * 64;
    const int lgrp = lane >> 4;
    const int lrow = lane & 15;

    // A staging: thread covers (row = t/2, k-halfrange = (t&1)*16 .. +16)
    const int arow = t >> 1;
    const int acol = (t & 1) * 16;
    // B staging: thread covers (k2 = (t>>4)*2 and k2+1, n-range = (t&15)*8 .. +8)
    const int k2   = (t >> 4) * 2;
    const int bn   = (t & 15) * 8;
    const bool bInB = !NGUARD || (n0 + bn) < Ncols;   // Ncols multiple of 8

    v8f acc[2][4] = {};
    unsigned int aSt[8], bSt[8];

    auto loadTile = [&](int k0) {
        // ---- A: 16 consecutive fp32 along K, 4x global_load_b128 ----
        const fvec4* ap = (const fvec4*)(A + (size_t)(m0 + arow) * lda + (k0 + acol));
        fvec4 av[4];
        #pragma unroll
        for (int v = 0; v < 4; ++v)
            av[v] = HAS_SCALE ? __builtin_nontemporal_load(&ap[v]) : ap[v];
        if (HAS_SCALE) {
            const fvec4* sp = (const fvec4*)(colScale + k0 + acol);
            #pragma unroll
            for (int v = 0; v < 4; ++v) av[v] *= sp[v];
        }
        #pragma unroll
        for (int v = 0; v < 4; ++v) {
            aSt[2 * v]     = pack_bf16x2(av[v][0], av[v][1]);
            aSt[2 * v + 1] = pack_bf16x2(av[v][2], av[v][3]);
        }
        // ---- B: rows k2, k2+1, 8 consecutive n each; 4x global_load_b128 ----
        fvec4 b0[2], b1[2];
        if (bInB) {
            const fvec4* bp0 = (const fvec4*)(B + (size_t)(k0 + k2) * ldb + n0 + bn);
            const fvec4* bp1 = (const fvec4*)(B + (size_t)(k0 + k2 + 1) * ldb + n0 + bn);
            b0[0] = bp0[0]; b0[1] = bp0[1];
            b1[0] = bp1[0]; b1[1] = bp1[1];
        } else {
            b0[0] = (fvec4)0.0f; b0[1] = (fvec4)0.0f;
            b1[0] = (fvec4)0.0f; b1[1] = (fvec4)0.0f;
        }
        #pragma unroll
        for (int e = 0; e < 8; ++e) bSt[e] = pack_bf16x2(b0[e >> 2][e & 3], b1[e >> 2][e & 3]);
    };

    auto storeTile = [&]() {
        #pragma unroll
        for (int v = 0; v < 8; ++v) Asm[arow][(acol >> 1) + v] = aSt[v];
        #pragma unroll
        for (int e = 0; e < 8; ++e) Bsm[bn + e][t >> 4] = bSt[e];
    };

    loadTile(kBeg);
    storeTile();
    __syncthreads();

    for (int k0 = kBeg; k0 < kEnd; k0 += 32) {
        const bool hasNext = (k0 + 32) < kEnd;
        if (hasNext) loadTile(k0 + 32);   // global loads in flight during WMMA

        // ---- fragments (ISA 16-bit A 16x32 / B 32x16 wave32 layouts) ----
        Frag af[2];
        #pragma unroll
        for (int i = 0; i < 2; ++i) {
            const uint4* p = (const uint4*)&Asm[wm + i * 16 + lrow][0];
            af[i].q[0] = p[lgrp];         // K 0..7   | 8..15
            af[i].q[1] = p[2 + lgrp];     // K 16..23 | 24..31
        }
        Frag bf[4];
        #pragma unroll
        for (int j = 0; j < 4; ++j) {
            const uint4* p = (const uint4*)&Bsm[wn + j * 16 + lrow][0];
            bf[j].q[0] = p[lgrp];         // K 0..7   | 8..15
            bf[j].q[1] = p[2 + lgrp];     // K 16..23 | 24..31
        }

        #pragma unroll
        for (int i = 0; i < 2; ++i)
            #pragma unroll
            for (int j = 0; j < 4; ++j)
                acc[i][j] = __builtin_amdgcn_wmma_f32_16x16x32_bf16(
                    false, af[i].v, false, bf[j].v,
                    (short)0, acc[i][j], false, false);

        __syncthreads();                  // everyone done reading this LDS tile
        if (hasNext) {
            storeTile();
            __syncthreads();
        }
    }

    // ---- epilogue ----
    float* Cz = Cmat + (size_t)blockIdx.z * (size_t)strideZ;
    #pragma unroll
    for (int i = 0; i < 2; ++i) {
        #pragma unroll
        for (int j = 0; j < 4; ++j) {
            const int gn = n0 + wn + j * 16 + lrow;
            if (NGUARD && gn >= Ncols) continue;
            const int gmBase = m0 + wm + i * 16 + lgrp * 8;
            const float bv = HAS_BIAS ? bias[gn] : 0.0f;
            #pragma unroll
            for (int r = 0; r < 8; ++r) {
                float v = acc[i][j][r] + bv;
                if (RELU) v = fmaxf(v, 0.0f);
                Cz[(size_t)(gmBase + r) * ldc + gn] = v;
            }
        }
    }
}

// ---------------------------------------------------------------------------
// Host-side launcher
// ---------------------------------------------------------------------------
extern "C" void kernel_launch(void* const* d_in, const int* in_sizes, int n_in,
                              void* d_out, int out_size, void* d_ws, size_t ws_size,
                              hipStream_t stream) {
    constexpr int N = 8192, F_IN = 1024, H = 512, C = 64;
    constexpr int NPART = 64;   // col-sum row chunks
    constexpr int KSPLIT = 4;   // split-K planes for final propagation

    const float* X     = (const float*)d_in[0];
    const float* tilde = (const float*)d_in[1];
    const float* W1    = (const float*)d_in[2];
    const float* b1    = (const float*)d_in[3];
    const float* W2    = (const float*)d_in[4];
    const float* b2    = (const float*)d_in[5];
    float* out = (float*)d_out;

    char* ws = (char*)d_ws;
    size_t off = 0;
    float* XW1   = (float*)(ws + off); off += (size_t)N * H * sizeof(float);           // 16 MB
    float* Hbuf  = (float*)(ws + off); off += (size_t)N * H * sizeof(float);           // 16 MB
    float* HW2   = (float*)(ws + off); off += (size_t)N * C * sizeof(float);           //  2 MB
    float* KPart = (float*)(ws + off); off += (size_t)KSPLIT * N * C * sizeof(float);  //  8 MB
    float* CPart = (float*)(ws + off); off += (size_t)NPART * N * sizeof(float);       //  2 MB
    float* D2    = (float*)(ws + off); off += (size_t)N * sizeof(float);
    float* scl   = (float*)(ws + off); off += (size_t)N * sizeof(float);

    // 1) normalization: D2 (rows), D1 (cols, two-level), scale
    rowsum_kernel<<<N, 256, 0, stream>>>(tilde, D2, N);
    colsum_part_kernel<<<dim3(N / 256, NPART), 256, 0, stream>>>(tilde, CPart, N, N / NPART);
    colsum_scale_kernel<<<N / 256, 256, 0, stream>>>(CPart, D2, scl, N, NPART);

    // 2) XW1 = X @ W1 + b1                 [N, H]
    gemm_bf16_wmma<false, true, false, false><<<dim3(H / 128, N / 128, 1), 256, 0, stream>>>(
        X, F_IN, W1, H, XW1, H, H, F_IN, F_IN, 0, nullptr, b1);

    // 3) Hbuf = relu(A_scaled @ XW1)       [N, H]
    gemm_bf16_wmma<true, false, true, false><<<dim3(H / 128, N / 128, 1), 256, 0, stream>>>(
        tilde, N, XW1, H, Hbuf, H, H, N, N, 0, scl, nullptr);

    // 4) HW2 = Hbuf @ W2 + b2              [N, C]
    gemm_bf16_wmma<false, true, false, true><<<dim3(1, N / 128, 1), 256, 0, stream>>>(
        Hbuf, H, W2, C, HW2, C, C, H, H, 0, nullptr, b2);

    // 5) out = A_scaled @ HW2              [N, C]  (split-K x4, then reduce)
    gemm_bf16_wmma<true, false, false, true><<<dim3(1, N / 128, KSPLIT), 256, 0, stream>>>(
        tilde, N, HW2, C, KPart, C, C, N, N / KSPLIT, (long long)N * C, scl, nullptr);
    splitk_reduce_kernel<<<(N * C + 255) / 256, 256, 0, stream>>>(
        KPart, out, N * C, KSPLIT, (size_t)N * C);

    (void)in_sizes; (void)n_in; (void)out_size; (void)ws_size;
}
